// Attention_29472065585724
// MI455X (gfx1250) — matrix operand order repaired and checked by reading, not compiled, not run on hardware
//
#include <hip/hip_runtime.h>
#include <hip/hip_bf16.h>

// ---------------------------------------------------------------------------
// Attention block for MI455X (gfx1250, wave32, WMMA bf16 w/ fp32 accumulate)
//   x:[16,1024,32,32] f32, w_qkv:[3072,1024], b_qkv:[3072],
//   w_proj:[1024,1024], b_proj:[1024]  -> out:[16,1024,32,32] f32
//
// Pipeline:
//   1) convert weights f32->bf16; convert+transpose x -> x_t[b][pix][c]
//   2) QKV GEMM (W[3072,1024] x X_b, tiles async-copied to LDS, double buffer)
//   3) S = scale*Q*K^T + softmax -> P  (WMMA, async staging)
//   4) O = P*V (WMMA), written transposed o_t[b][pix][c]
//   5) proj GEMM -> f32 output
// ---------------------------------------------------------------------------

typedef __bf16 bf16;
typedef __attribute__((ext_vector_type(16))) __bf16 v16bf;
typedef __attribute__((ext_vector_type(8)))  __bf16 v8bf;
typedef __attribute__((ext_vector_type(8)))  float  v8f;
typedef __attribute__((ext_vector_type(4)))  int    i32x4;

#define LDS_K 40   // 32 K-elems + 8 pad (16B-aligned row stride = 80B)

static constexpr int BATCH = 16;
static constexpr int C_IN  = 1024;
static constexpr int HW    = 1024;   // 32*32
static constexpr int SEQ   = 256;    // patch^2
static constexpr int NH    = 8;
static constexpr int HD    = 512;
static constexpr float ATT_SCALE = 0.08838834764831845f; // (1024/8)^-0.5

// ---------------------------------------------------------------------------
// async global->LDS copy path (CDNA5 GLOBAL_LOAD_ASYNC_TO_LDS_B128, ASYNCcnt)
// with a synchronous VGPR-staged fallback if the builtin is unavailable.
// ---------------------------------------------------------------------------
#if defined(__AMDGCN__) && __has_builtin(__builtin_amdgcn_global_load_async_to_lds_b128)
#define HAS_ASYNC_LDS 1
#else
#define HAS_ASYNC_LDS 0
#endif

__device__ __forceinline__ void cp_b128(bf16* lds_dst, const bf16* gsrc) {
#if HAS_ASYNC_LDS
  __builtin_amdgcn_global_load_async_to_lds_b128(
      (__attribute__((address_space(1))) i32x4*)gsrc,
      (__attribute__((address_space(3))) i32x4*)lds_dst,
      /*offset=*/0, /*cpol=*/0);
#else
  *(v8bf*)lds_dst = *(const v8bf*)gsrc;
#endif
}

template <int N>
__device__ __forceinline__ void cp_wait() {
#if HAS_ASYNC_LDS
#if __has_builtin(__builtin_amdgcn_s_wait_asynccnt)
  __builtin_amdgcn_s_wait_asynccnt(N);
#else
  asm volatile("s_wait_asynccnt %0" ::"i"(N) : "memory");
#endif
#endif
}

// ---------------------------------------------------------------------------
// addressing helpers for the channel/spatial-mixing reshape
// ---------------------------------------------------------------------------
// row offset (elements) of (q|k|v)[b,h,s,:] inside qkv buffer [b][3072][1024]
__device__ __forceinline__ size_t qkv_row_off(int b, int s, int t, int h) {
  int idx = s * 24 + t * 8 + h;
  int c   = idx >> 1;
  int pb  = idx & 1;
  return ((size_t)(b * 3072 + c) << 10) + ((size_t)pb << 9);
}

// ---------------------------------------------------------------------------
// WMMA fragment helpers (per ISA 16-bit A / B VGPR layouts)
// ---------------------------------------------------------------------------
__device__ __forceinline__ v16bf load_frag_a(const bf16* row, int kb) {
  v8bf lo = *(const v8bf*)(row + kb);        // K kb..kb+7
  v8bf hi = *(const v8bf*)(row + kb + 16);   // K kb+16..kb+23
  return __builtin_shufflevector(lo, hi, 0,1,2,3,4,5,6,7,8,9,10,11,12,13,14,15);
}

__device__ __forceinline__ v16bf load_frag_b(const bf16* row, int half) {
  const bf16* p = row + half * 8;            // K 8h..8h+7, 16+8h..16+8h+7
  v8bf lo = *(const v8bf*)(p);
  v8bf hi = *(const v8bf*)(p + 16);
  return __builtin_shufflevector(lo, hi, 0,1,2,3,4,5,6,7,8,9,10,11,12,13,14,15);
}

__device__ __forceinline__ v8f wmma_bf16(v16bf a, v16bf b, v8f c) {
  return __builtin_amdgcn_wmma_f32_16x16x32_bf16(
      false, a, false, b, (short)0, c, false, false);
}

// ---------------------------------------------------------------------------
// f32 -> bf16 elementwise convert (weights)
// ---------------------------------------------------------------------------
__global__ void cvt_f32_bf16_kernel(const float* __restrict__ in,
                                    bf16* __restrict__ out, int n) {
  int i = blockIdx.x * blockDim.x + threadIdx.x;
  if (i < n) out[i] = (bf16)in[i];
}

// ---------------------------------------------------------------------------
// f32 [b][R][C] -> bf16 [b][C][R] convert + transpose (32x32 LDS tiles)
// ---------------------------------------------------------------------------
__global__ void cvt_transpose_kernel(const float* __restrict__ in,
                                     bf16* __restrict__ out, int R, int C) {
  const int b  = blockIdx.z;
  const int c0 = blockIdx.x * 32;
  const int r0 = blockIdx.y * 32;
  __shared__ float tile[32][33];
  const int tx = threadIdx.x & 31;
  const int ty = threadIdx.x >> 5;   // 0..7
  const float* inb = in + (size_t)b * R * C;
  bf16* outb = out + (size_t)b * R * C;
#pragma unroll
  for (int p = 0; p < 4; ++p)
    tile[ty + p * 8][tx] = inb[(size_t)(r0 + ty + p * 8) * C + c0 + tx];
  __syncthreads();
#pragma unroll
  for (int p = 0; p < 4; ++p)
    outb[(size_t)(c0 + ty + p * 8) * R + r0 + tx] = (bf16)tile[tx][ty + p * 8];
}

// ---------------------------------------------------------------------------
// Batched GEMM + bias:  out_b[M,N] = W[M,K] * XT_b[N,K]^T + bias[M]
// Both operands row-contiguous in K -> pure b128 async staging, double buffer.
// 128x128 tile, 256 threads (8 waves), K-step 32.
// ---------------------------------------------------------------------------
template <typename OutT>
__global__ void __launch_bounds__(256)
gemm_bf16_kernel(const bf16* __restrict__ W,
                 const bf16* __restrict__ XT,
                 const float* __restrict__ bias,
                 OutT* __restrict__ out,
                 int M, int N, int K) {
  const int b = blockIdx.z;
  const bf16* XTb = XT + (size_t)b * N * K;
  OutT*      outb = out + (size_t)b * M * N;
  const int m0 = blockIdx.y * 128;
  const int n0 = blockIdx.x * 128;

  __shared__ bf16 As[2][128 * LDS_K];
  __shared__ bf16 Bs[2][128 * LDS_K];

  const int t    = threadIdx.x;
  const int lane = t & 31;
  const int wave = t >> 5;
  const int wm   = wave & 1;   // 2 wave rows -> 64 M-rows each
  const int wn   = wave >> 1;  // 4 wave cols -> 32 N-cols each
  const int row  = lane & 15;
  const int half = lane >> 4;
  const int kb   = half * 8;

  // per-thread staging coords: 512 x 16B chunks per tile, 2 per thread each
  const int sr0 = t >> 2;            // rows 0..63
  const int sc  = (t & 3) * 8;       // K-chunk 0..3

  auto stage = [&](int buf, int k0) {
#pragma unroll
    for (int i = 0; i < 2; ++i) {
      int r = sr0 + i * 64;
      cp_b128(&As[buf][r * LDS_K + sc], &W[(size_t)(m0 + r) * K + k0 + sc]);
    }
#pragma unroll
    for (int i = 0; i < 2; ++i) {
      int n = sr0 + i * 64;
      cp_b128(&Bs[buf][n * LDS_K + sc], &XTb[(size_t)(n0 + n) * K + k0 + sc]);
    }
  };

  v8f acc[4][2] = {};
  const int T = K / 32;
  stage(0, 0);

  for (int it = 0; it < T; ++it) {
    const int cur = it & 1;
    if (it + 1 < T) {
      stage(cur ^ 1, (it + 1) * 32);
      cp_wait<4>();          // tile `it`'s 4 async copies retired
    } else {
      cp_wait<0>();
    }
    __syncthreads();

    v16bf af[4], bfr[2];
#pragma unroll
    for (int i = 0; i < 4; ++i)
      af[i] = load_frag_a(&As[cur][(wm * 64 + i * 16 + row) * LDS_K], kb);
#pragma unroll
    for (int j = 0; j < 2; ++j)
      bfr[j] = load_frag_b(&Bs[cur][(wn * 32 + j * 16 + row) * LDS_K], half);
#pragma unroll
    for (int i = 0; i < 4; ++i)
#pragma unroll
      for (int j = 0; j < 2; ++j) acc[i][j] = wmma_bf16(af[i], bfr[j], acc[i][j]);
    __syncthreads();
  }

  const int col = lane & 15;
  const int rh  = half * 8;
#pragma unroll
  for (int i = 0; i < 4; ++i)
#pragma unroll
    for (int j = 0; j < 2; ++j) {
      int mb = m0 + wm * 64 + i * 16 + rh;
      int nb = n0 + wn * 32 + j * 16 + col;
#pragma unroll
      for (int r = 0; r < 8; ++r) {
        int m = mb + r;
        outb[(size_t)m * N + nb] = (OutT)(acc[i][j][r] + bias[m]);
      }
    }
}

// ---------------------------------------------------------------------------
// Attention pass 1: per (b, h, 64-row q-block) -> S = scale*Q*K^T, softmax,
// store P as bf16 [b][h][256][256].  Async staging, double buffered with
// ARITHMETIC buffer offsets (no pointer arrays -> no stack, LDS provable).
// Layout in smem (bytes):  [Qs0 5120 | Ks0 20480 | Qs1 5120 | Ks1 20480]
// then reused as S = 64x256 f32 (64KB) for the softmax.
// ---------------------------------------------------------------------------
#define QK_BUF_ELEMS 12800   // (5120+20480)/2 : bf16 elems per double-buffer slot

__global__ void __launch_bounds__(256)
attn_qk_softmax_kernel(const bf16* __restrict__ qkv, bf16* __restrict__ P) {
  const int qb = blockIdx.x;   // 0..3
  const int h  = blockIdx.y;   // 0..7
  const int b  = blockIdx.z;   // 0..15
  const int s0 = qb * 64;

  __shared__ __align__(16) unsigned char smem[64 * 256 * 4];
  bf16*  QS0 = (bf16*)smem;          // + cur*QK_BUF_ELEMS -> Qs ; +2560 -> Ks
  float* S   = (float*)smem;

  const int t    = threadIdx.x;
  const int lane = t & 31;
  const int wave = t >> 5;
  const int wm   = wave & 3;   // 4 row tiles (16 q rows each)
  const int wn   = wave >> 2;  // 2 col groups of 128 keys
  const int row  = lane & 15;
  const int half = lane >> 4;
  const int kb   = half * 8;

  const int sr0 = t >> 2;
  const int sc  = (t & 3) * 8;

  auto stage = [&](int buf, int kk) {
    bf16* Qs = QS0 + buf * QK_BUF_ELEMS;
    bf16* Ks = Qs + 2560;            // 5120 bytes after Qs
    cp_b128(&Qs[sr0 * LDS_K + sc],
            &qkv[qkv_row_off(b, s0 + sr0, 0, h) + kk + sc]);
#pragma unroll
    for (int i = 0; i < 4; ++i) {
      int r = sr0 + i * 64;
      cp_b128(&Ks[r * LDS_K + sc],
              &qkv[qkv_row_off(b, r, 1, h) + kk + sc]);
    }
  };

  v8f acc[8] = {};
  stage(0, 0);
  const int T = HD / 32;   // 16

  for (int it = 0; it < T; ++it) {
    const int cur = it & 1;
    if (it + 1 < T) {
      stage(cur ^ 1, (it + 1) * 32);
      cp_wait<5>();
    } else {
      cp_wait<0>();
    }
    __syncthreads();

    const bf16* Qs = QS0 + cur * QK_BUF_ELEMS;
    const bf16* Ks = Qs + 2560;
    v16bf a = load_frag_a(&Qs[(wm * 16 + row) * LDS_K], kb);
#pragma unroll
    for (int j = 0; j < 8; ++j) {
      v16bf bfr = load_frag_b(&Ks[(wn * 128 + j * 16 + row) * LDS_K], half);
      acc[j] = wmma_bf16(a, bfr, acc[j]);
    }
    __syncthreads();
  }

  // scatter scaled logits into LDS (staging dead after last barrier)
  const int col = lane & 15;
  const int rh  = half * 8;
#pragma unroll
  for (int j = 0; j < 8; ++j)
#pragma unroll
    for (int r = 0; r < 8; ++r) {
      int m = wm * 16 + rh + r;
      int n = wn * 128 + j * 16 + col;
      S[m * 256 + n] = acc[j][r] * ATT_SCALE;
    }
  __syncthreads();

  // row softmax (64 rows, one thread per row) + bf16 store of P
  if (t < 64) {
    float* srow = &S[t * 256];
    float mx = -3.402823466e+38f;
#pragma unroll 8
    for (int k = 0; k < 256; ++k) mx = fmaxf(mx, srow[k]);
    float sum = 0.f;
#pragma unroll 8
    for (int k = 0; k < 256; ++k) {
      float e = __expf(srow[k] - mx);
      srow[k] = e;
      sum += e;
    }
    float inv = __frcp_rn(sum);
    bf16* prow = P + (((size_t)(b * NH + h) * SEQ) + s0 + t) * SEQ;
#pragma unroll 8
    for (int k = 0; k < 256; ++k) prow[k] = (bf16)(srow[k] * inv);
  }
}

// ---------------------------------------------------------------------------
// Attention pass 2: O = P * V, written TRANSPOSED into o_t[b][pix][c] so the
// projection GEMM can stage its B tiles contiguously.
// workgroup = (b, h, 64-row m-block, 128-col n-block); K = 256 keys
// ---------------------------------------------------------------------------
__global__ void __launch_bounds__(256)
attn_pv_kernel(const bf16* __restrict__ qkv,
               const bf16* __restrict__ P,
               bf16* __restrict__ OT) {
  const int mb = blockIdx.x >> 2;
  const int nb = blockIdx.x & 3;
  const int h  = blockIdx.y;
  const int b  = blockIdx.z;
  const int s0 = mb * 64;
  const int d0 = nb * 128;

  __shared__ bf16 Ps[64 * LDS_K];
  __shared__ bf16 Vs[128 * LDS_K];   // V^T tile: Vs[d][k]

  const int t    = threadIdx.x;
  const int lane = t & 31;
  const int wave = t >> 5;
  const int wm   = wave & 3;   // 4 row tiles
  const int wn   = wave >> 2;  // 2 col groups of 64 dims
  const int row  = lane & 15;
  const int half = lane >> 4;
  const int kb   = half * 8;

  v8f acc[4] = {};
  const bf16* Pbase = P + ((size_t)(b * NH + h) * SEQ) * SEQ;

  for (int kk = 0; kk < SEQ; kk += 32) {
    {  // stage P block 64x32 (async-capable contiguous copy)
      int r  = t >> 2;
      int c8 = (t & 3) * 8;
      cp_b128(&Ps[r * LDS_K + c8], &Pbase[(size_t)(s0 + r) * SEQ + kk + c8]);
    }
#pragma unroll
    for (int i = 0; i < 2; ++i) {  // stage V transposed: V[kk+k][d0+d] -> Vs[d][k]
      int id = t + i * 256;
      int k  = id >> 4;
      int d8 = (id & 15) * 8;
      v8bf v = *(const v8bf*)&qkv[qkv_row_off(b, kk + k, 2, h) + d0 + d8];
#pragma unroll
      for (int e = 0; e < 8; ++e) Vs[(d8 + e) * LDS_K + k] = v[e];
    }
    cp_wait<0>();
    __syncthreads();

    v16bf a = load_frag_a(&Ps[(wm * 16 + row) * LDS_K], kb);
#pragma unroll
    for (int j = 0; j < 4; ++j) {
      v16bf bfr = load_frag_b(&Vs[(wn * 64 + j * 16 + row) * LDS_K], half);
      acc[j] = wmma_bf16(a, bfr, acc[j]);
    }
    __syncthreads();
  }

  const int col = lane & 15;
  const int rh  = half * 8;
#pragma unroll
  for (int j = 0; j < 4; ++j)
#pragma unroll
    for (int r = 0; r < 8; ++r) {
      int s   = s0 + wm * 16 + rh + r;
      int d   = d0 + wn * 64 + j * 16 + col;
      int idx = s * 8 + h;
      int c   = idx >> 1;
      int pix = ((idx & 1) << 9) + d;
      OT[((size_t)(b * 1024 + pix) << 10) + c] = (bf16)acc[j][r];
    }
}

// ---------------------------------------------------------------------------
// host launcher
// ---------------------------------------------------------------------------
extern "C" void kernel_launch(void* const* d_in, const int* in_sizes, int n_in,
                              void* d_out, int out_size, void* d_ws, size_t ws_size,
                              hipStream_t stream) {
  (void)in_sizes; (void)n_in; (void)out_size; (void)ws_size;

  const float* x      = (const float*)d_in[0];  // [16,1024,32,32]
  const float* w_qkv  = (const float*)d_in[1];  // [3072,1024]
  const float* b_qkv  = (const float*)d_in[2];  // [3072]
  const float* w_proj = (const float*)d_in[3];  // [1024,1024]
  const float* b_proj = (const float*)d_in[4];  // [1024]
  float* out = (float*)d_out;

  const size_t n_x     = (size_t)BATCH * C_IN * HW;      // 16,777,216
  const size_t n_wqkv  = (size_t)3 * C_IN * C_IN;        //  3,145,728
  const size_t n_wproj = (size_t)C_IN * C_IN;            //  1,048,576
  const size_t n_qkv   = (size_t)BATCH * 3 * C_IN * HW;  // 50,331,648
  const size_t n_p     = (size_t)BATCH * NH * SEQ * SEQ; //  8,388,608

  unsigned char* ws = (unsigned char*)d_ws;
  bf16* x_t      = (bf16*)ws;  ws += n_x * 2;      // x transposed [b][pix][c]
  bf16* wqkv_bf  = (bf16*)ws;  ws += n_wqkv * 2;
  bf16* wproj_bf = (bf16*)ws;  ws += n_wproj * 2;
  bf16* qkv_bf   = (bf16*)ws;  ws += n_qkv * 2;    // [b][3072][1024]
  bf16* p_bf     = (bf16*)ws;  ws += n_p * 2;      // [b][h][256][256]
  bf16* o_t      = (bf16*)ws;                      // O transposed [b][pix][c]

  // 1) weight converts + x convert/transpose
  cvt_f32_bf16_kernel<<<(n_wqkv + 255) / 256, 256, 0, stream>>>(w_qkv, wqkv_bf, (int)n_wqkv);
  cvt_f32_bf16_kernel<<<(n_wproj + 255) / 256, 256, 0, stream>>>(w_proj, wproj_bf, (int)n_wproj);
  cvt_transpose_kernel<<<dim3(HW / 32, C_IN / 32, BATCH), 256, 0, stream>>>(x, x_t, C_IN, HW);

  // 2) QKV GEMM: qkv_b[3072,1024] = W_qkv * X_b + b_qkv   (bf16 out)
  gemm_bf16_kernel<bf16><<<dim3(HW / 128, 3072 / 128, BATCH), 256, 0, stream>>>(
      wqkv_bf, x_t, b_qkv, qkv_bf, 3 * C_IN, HW, C_IN);

  // 3) attention
  attn_qk_softmax_kernel<<<dim3(SEQ / 64, NH, BATCH), 256, 0, stream>>>(qkv_bf, p_bf);
  attn_pv_kernel<<<dim3((SEQ / 64) * (HD / 128), NH, BATCH), 256, 0, stream>>>(
      qkv_bf, p_bf, o_t);

  // 4) projection GEMM: out_b[1024,1024] = W_proj * O_b + b_proj   (f32 out)
  gemm_bf16_kernel<float><<<dim3(HW / 128, C_IN / 128, BATCH), 256, 0, stream>>>(
      wproj_bf, o_t, b_proj, out, C_IN, HW, C_IN);
}
